// HardVectorQuantizer_52321291600402
// MI455X (gfx1250) — matrix operand order repaired and checked by reading, not compiled, not run on hardware
//
#include <hip/hip_runtime.h>
#include <hip/hip_bf16.h>
#include <math.h>

// Problem constants (fixed by the reference): z is (4,2048,512) -> N=8192 rows,
// D=512, codebook K=8192.
#define VQ_N 8192
#define VQ_D 512
#define VQ_K 8192
#define VQ_KCHUNKS 8                 // K-split: 8 waves per block, one chunk each
#define VQ_KC (VQ_K / VQ_KCHUNKS)    // 1024 codes per wave

typedef __attribute__((ext_vector_type(16))) __bf16 v16bf;
typedef __attribute__((ext_vector_type(8)))  float  v8f;

static __device__ __forceinline__ unsigned short f2bf(float f) {
  unsigned int u = __float_as_uint(f);
  u += 0x7fffu + ((u >> 16) & 1u);          // RTNE
  return (unsigned short)(u >> 16);
}
static __device__ __forceinline__ float bf2f(unsigned short h) {
  return __uint_as_float(((unsigned int)h) << 16);
}

// ---------------------------------------------------------------------------
// Pack z (N x D f32 row-major) into WMMA A-fragment order, hi/lo bf16 split.
// A-fragment (16x32, ISA 7.12.2): lane l holds source row (l&15); bf16 element
// i (= 2*vgpr + half) holds k = i + (i<8 ? 0 : 8) + (l>=16 ? 8 : 0).
// Layout: [mtile][dc][lane][16] contiguous -> 32B aligned per-lane loads.
// Byte address of fragment (mtile, dc), lane l: mtile*16384 + dc*1024 + l*32.
// ---------------------------------------------------------------------------
__global__ void vq_pack_z(const float* __restrict__ z,
                          unsigned short* __restrict__ hi,
                          unsigned short* __restrict__ lo) {
  const int mtile = blockIdx.x;                      // N/16 tiles
  for (int p = threadIdx.x; p < 16 * 32; p += blockDim.x) {
    const int dc   = p >> 5;                         // 16 chunks of 32 along D
    const int lane = p & 31;
    const int row  = mtile * 16 + (lane & 15);
    const size_t outBase = ((size_t)(mtile * 16 + dc) * 32 + lane) * 16;
#pragma unroll
    for (int i = 0; i < 16; ++i) {
      const int k = i + ((i < 8) ? 0 : 8) + ((lane >= 16) ? 8 : 0);
      const float x = z[(size_t)row * VQ_D + dc * 32 + k];
      const unsigned short h = f2bf(x);
      hi[outBase + i] = h;
      lo[outBase + i] = f2bf(x - bf2f(h));
    }
  }
}

// ---------------------------------------------------------------------------
// Pack emb (K x D f32 row-major) into WMMA B-fragment order (B[k][n]=emb[n][k]):
// lane l holds emb row n=(l&15); element i holds k = i + (i<8 ? 0 : 8) + (l>=16 ? 8 : 0).
// ---------------------------------------------------------------------------
__global__ void vq_pack_e(const float* __restrict__ e,
                          unsigned short* __restrict__ hi,
                          unsigned short* __restrict__ lo) {
  const int ntile = blockIdx.x;                      // K/16 tiles
  for (int p = threadIdx.x; p < 16 * 32; p += blockDim.x) {
    const int dc   = p >> 5;
    const int lane = p & 31;
    const int row  = ntile * 16 + (lane & 15);
    const size_t outBase = ((size_t)(ntile * 16 + dc) * 32 + lane) * 16;
#pragma unroll
    for (int i = 0; i < 16; ++i) {
      const int k = i + ((i < 8) ? 0 : 8) + ((lane >= 16) ? 8 : 0);
      const float x = e[(size_t)row * VQ_D + dc * 32 + k];
      const unsigned short h = f2bf(x);
      hi[outBase + i] = h;
      lo[outBase + i] = f2bf(x - bf2f(h));
    }
  }
}

// ||e_k||^2, one wave32 per codebook row (coalesced), 8 waves per block.
__global__ void vq_norms(const float* __restrict__ e, float* __restrict__ norms) {
  const int row  = blockIdx.x * 8 + (threadIdx.x >> 5);
  const int lane = threadIdx.x & 31;
  float s = 0.0f;
  for (int d = lane; d < VQ_D; d += 32) {
    const float v = e[(size_t)row * VQ_D + d];
    s += v * v;
  }
#pragma unroll
  for (int m = 16; m > 0; m >>= 1) s += __shfl_xor(s, m, 32);
  if (lane == 0) norms[row] = s;
}

// ---------------------------------------------------------------------------
// Fused distance-GEMM + argmin. score(m,n) = ||e_n||^2 - 2*dot(z_m, e_n).
// Split-bf16 dot: hi*hi + hi*lo + lo*hi (3x v_wmma_f32_16x16x32_bf16 per term),
// ~2^-18 relative error (fp32-grade ranking).
//
// Block = 8 waves, owns TWO M-tiles (32 z-rows) staged once in LDS (64 KB) and
// shared by all waves. Each wave covers one 1024-code K-chunk, register-blocked
// 4 N-tiles x 2 M-tiles, WMMAs interleaved across accumulators.
//
// All hot-loop addresses are (single 32-bit base) + compile-time immediate.
// An opaque zero offset (inline asm) makes the LDS A addresses formally
// loop-variant so LICM cannot hoist 512 VGPRs worth of A out of the g loop
// (that caused scratch spills in the previous build); A is re-read from LDS
// each group, keeping the wave under the 256-VGPR / 4-waves-per-SIMD budget.
// ---------------------------------------------------------------------------
__global__ void __launch_bounds__(256, 4) vq_argmin_gemm(
    const unsigned short* __restrict__ zA_hi,
    const unsigned short* __restrict__ zA_lo,
    const unsigned short* __restrict__ eB_hi,
    const unsigned short* __restrict__ eB_lo,
    const float* __restrict__ norms,
    float* __restrict__ partVal,
    int*   __restrict__ partIdx) {
  const int lane   = threadIdx.x & 31;
  const int kchunk = threadIdx.x >> 5;          // 8 waves = 8 K-chunks
  const int mpair  = blockIdx.x;                // 256 blocks: mtiles 2b, 2b+1
  const int ntile0 = kchunk * (VQ_KC / 16);     // 64 N-tiles per chunk

  // 2 mtiles x 16 dc x 32 lanes x 16 bf16 = 16384 ushort per part (32 KB each)
  __shared__ unsigned short ldsAhi[2 * 16 * 32 * 16];
  __shared__ unsigned short ldsAlo[2 * 16 * 32 * 16];

  {
    // Packed zA is mtile-major: two consecutive mtiles are one contiguous
    // 32 KB region per part. Stage with b128 copies.
    const size_t srcBase = (size_t)mpair * (2 * 16 * 32 * 16);
    const uint4* gh = (const uint4*)(zA_hi + srcBase);
    const uint4* gl = (const uint4*)(zA_lo + srcBase);
    uint4* sh = (uint4*)ldsAhi;
    uint4* sl = (uint4*)ldsAlo;
    for (int i = threadIdx.x; i < 2048; i += 256) {   // 2048 uint4 per part
      sh[i] = gh[i];
      sl[i] = gl[i];
    }
  }
  __syncthreads();

  float minv[2][8];
  int   mini[2][8];
#pragma unroll
  for (int mt = 0; mt < 2; ++mt)
#pragma unroll
    for (int r = 0; r < 8; ++r) {
      minv[mt][r] = __builtin_huge_valf();
      mini[mt][r] = 0x7fffffff;
    }

  // Per-lane LDS byte bases (immediate offsets select mt/dc/half).
  const char* aHiB = (const char*)ldsAhi + lane * 32;
  const char* aLoB = (const char*)ldsAlo + lane * 32;

  for (int g = 0; g < (VQ_KC / 64); ++g) {      // 16 groups of 4 N-tiles
    const int ntb = ntile0 + g * 4;
    // Single per-lane 32-bit byte offset for this group's B fragments.
    const unsigned bBase = (unsigned)ntb * 16384u + (unsigned)lane * 32u;
    const char* bHiB = (const char*)eB_hi + bBase;
    const char* bLoB = (const char*)eB_lo + bBase;

    // Opaque zero: defeats LICM/CSE of the LDS A loads across g iterations
    // while keeping the address rooted at the __shared__ arrays (still ds_*).
    unsigned opq = 0;
    asm volatile("" : "+v"(opq));
    const char* aH = aHiB + opq;
    const char* aL = aLoB + opq;

    v8f C[4][2] = {};                           // [ntile][mtile]
#pragma unroll
    for (int dc = 0; dc < 16; ++dc) {
      const v16bf a0h = *(const v16bf*)(aH + dc * 1024);
      const v16bf a0l = *(const v16bf*)(aL + dc * 1024);
      const v16bf a1h = *(const v16bf*)(aH + 16384 + dc * 1024);
      const v16bf a1l = *(const v16bf*)(aL + 16384 + dc * 1024);
#pragma unroll
      for (int t = 0; t < 4; ++t) {
        const v16bf bH = *(const v16bf*)(bHiB + (t * 16384 + dc * 1024));
        const v16bf bL = *(const v16bf*)(bLoB + (t * 16384 + dc * 1024));
        // Interleaved: consecutive WMMAs target different accumulators.
        C[t][0] = __builtin_amdgcn_wmma_f32_16x16x32_bf16(false, a0h, false, bH,
                                                          (short)0, C[t][0], false, false);
        C[t][1] = __builtin_amdgcn_wmma_f32_16x16x32_bf16(false, a1h, false, bH,
                                                          (short)0, C[t][1], false, false);
        C[t][0] = __builtin_amdgcn_wmma_f32_16x16x32_bf16(false, a0h, false, bL,
                                                          (short)0, C[t][0], false, false);
        C[t][1] = __builtin_amdgcn_wmma_f32_16x16x32_bf16(false, a1h, false, bL,
                                                          (short)0, C[t][1], false, false);
        C[t][0] = __builtin_amdgcn_wmma_f32_16x16x32_bf16(false, a0l, false, bH,
                                                          (short)0, C[t][0], false, false);
        C[t][1] = __builtin_amdgcn_wmma_f32_16x16x32_bf16(false, a1l, false, bH,
                                                          (short)0, C[t][1], false, false);
      }
    }
    // C layout: VGPR r, lane l -> row M = r + (l>=16?8:0), col N = l&15.
#pragma unroll
    for (int t = 0; t < 4; ++t) {
      const int n = (ntb + t) * 16 + (lane & 15);
      const float nrm = norms[n];
#pragma unroll
      for (int mt = 0; mt < 2; ++mt)
#pragma unroll
        for (int r = 0; r < 8; ++r) {
          const float val = nrm - 2.0f * C[t][mt][r];
          if (val < minv[mt][r]) { minv[mt][r] = val; mini[mt][r] = n; }
        }
    }
  }

  // Reduce across the 16 lanes sharing a row; tie-break on smaller index to
  // match jnp.argmin (first minimum).
#pragma unroll
  for (int mt = 0; mt < 2; ++mt)
#pragma unroll
    for (int r = 0; r < 8; ++r) {
      float v = minv[mt][r];
      int   ix = mini[mt][r];
#pragma unroll
      for (int m = 1; m < 16; m <<= 1) {
        const float ov = __shfl_xor(v, m, 32);
        const int   oi = __shfl_xor(ix, m, 32);
        if (ov < v || (ov == v && oi < ix)) { v = ov; ix = oi; }
      }
      if ((lane & 15) == 0) {
        const int row = (mpair * 2 + mt) * 16 + r + ((lane >> 4) << 3);
        partVal[(size_t)row * VQ_KCHUNKS + kchunk] = v;
        partIdx[(size_t)row * VQ_KCHUNKS + kchunk] = ix;
      }
    }
}

// Combine K-chunk partials -> final argmin per row (chunks are in ascending
// code ranges, so strict < keeps the first minimum).
__global__ void vq_argmin_reduce(const float* __restrict__ partVal,
                                 const int* __restrict__ partIdx,
                                 int* __restrict__ finalIdx) {
  const int row = blockIdx.x * blockDim.x + threadIdx.x;
  if (row >= VQ_N) return;
  float bv = __builtin_huge_valf();
  int   bi = 0;
  for (int c = 0; c < VQ_KCHUNKS; ++c) {
    const float v = partVal[(size_t)row * VQ_KCHUNKS + c];
    const int  ix = partIdx[(size_t)row * VQ_KCHUNKS + c];
    if (v < bv) { bv = v; bi = ix; }
  }
  finalIdx[row] = bi;
}

// Gather z_q, residual, per-row sum((z-z_q)^2)*valid, and histogram counts.
__global__ void vq_gather(const float* __restrict__ z,
                          const float* __restrict__ emb,
                          const unsigned char* __restrict__ mask,
                          const int* __restrict__ finalIdx,
                          float* __restrict__ outRes,
                          float* __restrict__ outZq,
                          float* __restrict__ rowSq,
                          unsigned int* __restrict__ counts) {
  const int row  = blockIdx.x;
  const int idx  = finalIdx[row];
  const int valid = (mask[row] == 0) ? 1 : 0;
  float s = 0.0f;
  for (int d = threadIdx.x; d < VQ_D; d += blockDim.x) {
    const float zq = emb[(size_t)idx * VQ_D + d];
    const float zz = z[(size_t)row * VQ_D + d];
    outZq[(size_t)row * VQ_D + d]  = zq;
    const float diff = zz - zq;
    outRes[(size_t)row * VQ_D + d] = diff;
    s += diff * diff;
  }
  s *= (float)valid;
  __shared__ float red[256];
  red[threadIdx.x] = s;
  __syncthreads();
  for (int st = 128; st > 0; st >>= 1) {
    if (threadIdx.x < st) red[threadIdx.x] += red[threadIdx.x + st];
    __syncthreads();
  }
  if (threadIdx.x == 0) {
    rowSq[row] = red[0];
    if (valid) atomicAdd(&counts[idx], 1u);    // integer atomic: deterministic
  }
}

// Scalars: vq_loss, perplexity, codes_used, codes_used_frac.
__global__ void vq_finalize(const unsigned int* __restrict__ counts,
                            const float* __restrict__ rowSq,
                            float* __restrict__ outLoss,
                            float* __restrict__ outPerp,
                            float* __restrict__ outUsed,
                            float* __restrict__ outFrac) {
  __shared__ float    fred[256];
  __shared__ unsigned ured[256];
  __shared__ float    s_total, s_invTot;
  const int tid = threadIdx.x;

  unsigned t = 0;
  for (int k = tid; k < VQ_K; k += 256) t += counts[k];
  ured[tid] = t;
  __syncthreads();
  for (int st = 128; st > 0; st >>= 1) {
    if (tid < st) ured[tid] += ured[tid + st];
    __syncthreads();
  }
  if (tid == 0) {
    unsigned tot = ured[0];
    if (tot == 0) tot = 1;                    // max(total,1); also n_valid
    s_total  = (float)tot;
    s_invTot = 1.0f / (float)tot;
  }
  __syncthreads();
  const float invTot = s_invTot;

  float ent = 0.0f, used = 0.0f;
  for (int k = tid; k < VQ_K; k += 256) {
    const unsigned c = counts[k];
    const float p = (float)c * invTot;
    ent -= p * logf(p + 1e-10f);
    if (c > 0) used += 1.0f;
  }
  fred[tid] = ent;
  __syncthreads();
  for (int st = 128; st > 0; st >>= 1) {
    if (tid < st) fred[tid] += fred[tid + st];
    __syncthreads();
  }
  const float entropy = fred[0];
  __syncthreads();

  fred[tid] = used;
  __syncthreads();
  for (int st = 128; st > 0; st >>= 1) {
    if (tid < st) fred[tid] += fred[tid + st];
    __syncthreads();
  }
  const float usedTot = fred[0];
  __syncthreads();

  float sq = 0.0f;
  for (int r = tid; r < VQ_N; r += 256) sq += rowSq[r];
  fred[tid] = sq;
  __syncthreads();
  for (int st = 128; st > 0; st >>= 1) {
    if (tid < st) fred[tid] += fred[tid + st];
    __syncthreads();
  }
  if (tid == 0) {
    const float denom = s_total * (float)VQ_D;
    // loss_codebook == loss_commit numerically -> (1 + 0.25) * mse
    *outLoss = 1.25f * fred[0] / denom;
    *outPerp = expf(entropy);
    *outUsed = usedTot;
    *outFrac = usedTot / (float)VQ_K;
  }
}

extern "C" void kernel_launch(void* const* d_in, const int* in_sizes, int n_in,
                              void* d_out, int out_size, void* d_ws, size_t ws_size,
                              hipStream_t stream) {
  const float*         z    = (const float*)d_in[0];          // N x D
  const unsigned char* mask = (const unsigned char*)d_in[1];  // N bools (all 0 here)
  const float*         emb  = (const float*)d_in[2];          // K x D

  float* out = (float*)d_out;
  const size_t ND = (size_t)in_sizes[0];    // 4,194,304
  float* outRes  = out;                     // z_residual
  float* outLoss = out + ND;                // vq_loss
  float* outZq   = out + ND + 1;            // z_q
  float* outPerp = out + 2 * ND + 1;        // perplexity
  float* outUsed = out + 2 * ND + 2;        // codes_used
  float* outFrac = out + 2 * ND + 3;        // codes_used_frac

  // Workspace carve-up (256B aligned), ~34.2 MB total.
  char* w = (char*)d_ws;
  size_t off = 0;
  auto carve = [&](size_t bytes) {
    void* p = w + off;
    off = (off + bytes + 255) & ~(size_t)255;
    return p;
  };
  unsigned short* zA_hi = (unsigned short*)carve((size_t)VQ_N * VQ_D * 2);
  unsigned short* zA_lo = (unsigned short*)carve((size_t)VQ_N * VQ_D * 2);
  unsigned short* eB_hi = (unsigned short*)carve((size_t)VQ_K * VQ_D * 2);
  unsigned short* eB_lo = (unsigned short*)carve((size_t)VQ_K * VQ_D * 2);
  float*        norms   = (float*)carve((size_t)VQ_K * 4);
  float*        partVal = (float*)carve((size_t)VQ_N * VQ_KCHUNKS * 4);
  int*          partIdx = (int*)carve((size_t)VQ_N * VQ_KCHUNKS * 4);
  int*          finalIdx= (int*)carve((size_t)VQ_N * 4);
  unsigned int* counts  = (unsigned int*)carve((size_t)VQ_K * 4);
  float*        rowSq   = (float*)carve((size_t)VQ_N * 4);

  hipMemsetAsync(counts, 0, (size_t)VQ_K * 4, stream);

  vq_pack_z<<<VQ_N / 16, 256, 0, stream>>>(z, zA_hi, zA_lo);
  vq_pack_e<<<VQ_K / 16, 256, 0, stream>>>(emb, eB_hi, eB_lo);
  vq_norms <<<VQ_K / 8,  256, 0, stream>>>(emb, norms);

  vq_argmin_gemm<<<VQ_N / 32, 256, 0, stream>>>(zA_hi, zA_lo, eB_hi, eB_lo,
                                                norms, partVal, partIdx);
  vq_argmin_reduce<<<VQ_N / 256, 256, 0, stream>>>(partVal, partIdx, finalIdx);
  vq_gather<<<VQ_N, 256, 0, stream>>>(z, emb, mask, finalIdx,
                                      outRes, outZq, rowSq, counts);
  vq_finalize<<<1, 256, 0, stream>>>(counts, rowSq, outLoss, outPerp, outUsed, outFrac);
}
